// GNNBackbone_77610059039208
// MI455X (gfx1250) — matrix operand order repaired and checked by reading, not compiled, not run on hardware
//
#include <hip/hip_runtime.h>
#include <hip/hip_bf16.h>
#include <math.h>

typedef __attribute__((ext_vector_type(16))) __bf16 v16bf;
typedef __attribute__((ext_vector_type(8)))  float  v8f;

#define NN 50000
#define EE 200000
#define EP (EE + NN)
#define BN_EPS 1e-5f
#define NEG_SLOPE 0.2f

// ---------- helpers ----------
__device__ __forceinline__ unsigned int f2bf_bits(float f) {
    union { float f; unsigned int u; } x; x.f = f;
    unsigned int r = x.u + 0x7FFFu + ((x.u >> 16) & 1u);
    return r >> 16;
}
__device__ __forceinline__ unsigned int pack2bf(float lo, float hi) {
    return f2bf_bits(lo) | (f2bf_bits(hi) << 16);
}
__device__ __forceinline__ __bf16 bits2bf(unsigned short s) {
    union { unsigned short s; __bf16 b; } x; x.s = s; return x.b;
}
__device__ __forceinline__ float lrelu(float x) { return x > 0.f ? x : NEG_SLOPE * x; }

__device__ __forceinline__ void edge_sd(const int* __restrict__ ei, int e, int& s, int& d) {
    if (e < EE) { s = ei[e]; d = ei[EE + e]; }
    else        { s = e - EE; d = e - EE; }
}

__device__ __forceinline__ void atomicMaxF(float* addr, float val) {
    if (val >= 0.f) atomicMax((int*)addr, __float_as_int(val));
    else            atomicMin((unsigned int*)addr, (unsigned int)__float_as_int(val));
}

// ---------- fill ----------
__global__ void fill_f32(float* __restrict__ p, float v, int n) {
    int i = blockIdx.x * blockDim.x + threadIdx.x;
    if (i < n) p[i] = v;
}

// ---------- WMMA GEMM: C[M,N] = A[M,K] @ B[K,N] (fp32 in, bf16 WMMA, fp32 out) ----------
// block = 256 threads (8 waves), block tile 128x64, K step 32
__global__ void __launch_bounds__(256)
wmma_gemm(const float* __restrict__ A, const float* __restrict__ B,
          float* __restrict__ C, int M, int K, int N,
          const float* __restrict__ bias, int do_relu) {
    __shared__ unsigned short As[128][34];   // [row][k]  (row-major, 4B-aligned rows+16-col halves)
    __shared__ unsigned short BsT[64][34];   // [col][k]  (transposed so operand reads are contiguous)

    const int tid  = threadIdx.x;
    const int lane = tid & 31;
    const int wid  = tid >> 5;
    const int m0 = blockIdx.x * 128;
    const int n0 = blockIdx.y * 64;

    const int lrow = lane & 15;
    const int lhi  = lane >> 4;          // 0 or 1

    v8f acc0 = {}, acc1 = {}, acc2 = {}, acc3 = {};

    // A-load mapping: 256 thr * 16 elems = 128x32; clamp row (OOB rows feed only
    // guarded epilogue rows, so duplicated data is harmless and loads stay vectorized)
    const int a_row = tid >> 1;
    const int a_c0  = (tid & 1) * 16;
    int a_grow = m0 + a_row; if (a_grow >= M) a_grow = M - 1;
    const float4* a_src = (const float4*)(A + (size_t)a_grow * K + a_c0);

    // B-load mapping: 256 thr * 8 elems = 32x64
    const int b_k   = tid >> 3;
    const int b_c0  = (tid & 7) * 8;
    const float4* b_src = (const float4*)(B + (size_t)b_k * N + n0 + b_c0);
    const int b_kstride_f4 = (N * 32) >> 2;  // float4 step for kk += 32

    for (int kk = 0; kk < K; kk += 32) {
        // stage A: 4x float4 -> 8x packed-b32 LDS stores
        {
            unsigned int* dst = (unsigned int*)&As[a_row][a_c0];
            #pragma unroll
            for (int j = 0; j < 4; ++j) {
                float4 v = a_src[(kk >> 2) + j];
                dst[2 * j + 0] = pack2bf(v.x, v.y);
                dst[2 * j + 1] = pack2bf(v.z, v.w);
            }
        }
        // stage B transposed (strided b16 stores; reads are the hot side)
        {
            const float4* bg = b_src + (kk >> 5) * b_kstride_f4;
            #pragma unroll
            for (int j = 0; j < 2; ++j) {
                float4 v = bg[j];
                BsT[b_c0 + 4 * j + 0][b_k] = (unsigned short)f2bf_bits(v.x);
                BsT[b_c0 + 4 * j + 1][b_k] = (unsigned short)f2bf_bits(v.y);
                BsT[b_c0 + 4 * j + 2][b_k] = (unsigned short)f2bf_bits(v.z);
                BsT[b_c0 + 4 * j + 3][b_k] = (unsigned short)f2bf_bits(v.w);
            }
        }
        __syncthreads();

        // A operand: lanes 0-15 -> K base 0, lanes 16-31 -> K base 16;
        // elems 0..15 -> K=base+i
        v16bf a;
        {
            const int row  = wid * 16 + lrow;
            const int base = lhi * 16;
            #pragma unroll
            for (int i = 0; i < 8; ++i) {
                a[i]     = bits2bf(As[row][base + i]);
                a[i + 8] = bits2bf(As[row][base + 8 + i]);
            }
        }
        // B operand per 16-col tile: lanes 0-15 -> K=0..15, lanes 16-31 -> K=16..31
        const int koff = lhi * 16;
        #pragma unroll
        for (int nt = 0; nt < 4; ++nt) {
            v16bf b;
            const int col = nt * 16 + lrow;
            #pragma unroll
            for (int i = 0; i < 16; ++i) b[i] = bits2bf(BsT[col][koff + i]);
            if (nt == 0) acc0 = __builtin_amdgcn_wmma_f32_16x16x32_bf16(false, a, false, b, (short)0, acc0, false, false);
            if (nt == 1) acc1 = __builtin_amdgcn_wmma_f32_16x16x32_bf16(false, a, false, b, (short)0, acc1, false, false);
            if (nt == 2) acc2 = __builtin_amdgcn_wmma_f32_16x16x32_bf16(false, a, false, b, (short)0, acc2, false, false);
            if (nt == 3) acc3 = __builtin_amdgcn_wmma_f32_16x16x32_bf16(false, a, false, b, (short)0, acc3, false, false);
        }
        __syncthreads();
    }

    // epilogue: C/D layout: VGPR r -> M = r + 8*lhi, N = lane%16
    #pragma unroll
    for (int nt = 0; nt < 4; ++nt) {
        v8f acc = (nt == 0) ? acc0 : (nt == 1) ? acc1 : (nt == 2) ? acc2 : acc3;
        const int gcol = n0 + nt * 16 + lrow;
        float badd = bias ? bias[gcol] : 0.f;
        #pragma unroll
        for (int r = 0; r < 8; ++r) {
            const int grow = m0 + wid * 16 + lhi * 8 + r;
            if (grow < M) {
                float v = acc[r] + badd;
                if (do_relu) v = fmaxf(v, 0.f);
                C[(size_t)grow * N + gcol] = v;
            }
        }
    }
}

// ---------- per-(node,head) attention scores: wave32 reduction over 128 channels ----------
__global__ void attn_scores(const float* __restrict__ xp,
                            const float* __restrict__ asrc, const float* __restrict__ adst,
                            float* __restrict__ es, float* __restrict__ ed,
                            int heads, int n_items /* = N*heads */) {
    const int gid  = blockIdx.x * blockDim.x + threadIdx.x;
    const int idx  = gid >> 5;
    const int lane = gid & 31;
    if (idx >= n_items) return;
    const int node = idx / heads;
    const int h    = idx % heads;
    const float4* xv = (const float4*)(xp + (size_t)node * heads * 128 + h * 128 + lane * 4);
    const float4* av = (const float4*)(asrc + h * 128 + lane * 4);
    const float4* dv = (const float4*)(adst + h * 128 + lane * 4);
    float4 x = *xv, a = *av, d = *dv;
    float ss = x.x * a.x + x.y * a.y + x.z * a.z + x.w * a.w;
    float dd = x.x * d.x + x.y * d.y + x.z * d.z + x.w * d.w;
    #pragma unroll
    for (int off = 16; off > 0; off >>= 1) {
        ss += __shfl_down(ss, off, 32);
        dd += __shfl_down(dd, off, 32);
    }
    if (lane == 0) { es[idx] = ss; ed[idx] = dd; }
}

// ---------- edge pass 1: segment max ----------
__global__ void edge_max(const int* __restrict__ ei, const float* __restrict__ es,
                         const float* __restrict__ ed, float* __restrict__ m, int heads) {
    const int t = blockIdx.x * blockDim.x + threadIdx.x;
    if (t >= EP * heads) return;
    const int e = t / heads, h = t % heads;
    int s, d; edge_sd(ei, e, s, d);
    const float el = lrelu(es[s * heads + h] + ed[d * heads + h]);
    atomicMaxF(&m[d * heads + h], el);
}

// ---------- edge pass 2: w = exp(e - m[dst]); segment sum ----------
__global__ void edge_w(const int* __restrict__ ei, const float* __restrict__ es,
                       const float* __restrict__ ed, const float* __restrict__ m,
                       float* __restrict__ ssum, float* __restrict__ wbuf, int heads) {
    const int t = blockIdx.x * blockDim.x + threadIdx.x;
    if (t >= EP * heads) return;
    const int e = t / heads, h = t % heads;
    int s, d; edge_sd(ei, e, s, d);
    const float el = lrelu(es[s * heads + h] + ed[d * heads + h]);
    const float w = __expf(el - m[d * heads + h]);
    wbuf[t] = w;
    atomicAdd(&ssum[d * heads + h], w);
}

// ---------- edge pass 3: out[dst] += alpha * xp[src] ----------
__global__ void edge_agg(const int* __restrict__ ei, const float* __restrict__ xp,
                         const float* __restrict__ wbuf, const float* __restrict__ ssum,
                         float* __restrict__ outb, int heads, int ctot) {
    const int chunks = ctot >> 2;                  // float4 chunks per edge
    const int t = blockIdx.x * blockDim.x + threadIdx.x;
    if (t >= EP * chunks) return;
    const int e = t / chunks, q = t % chunks;
    const int h = q / 32;                          // 128 ch per head -> 32 chunks
    int s, d; edge_sd(ei, e, s, d);
    const float alpha = wbuf[e * heads + h] / (ssum[d * heads + h] + 1e-16f);
    const float4 xv = *(const float4*)(xp + (size_t)s * ctot + q * 4);
    float* o = outb + (size_t)d * ctot + q * 4;
    atomicAdd(o + 0, alpha * xv.x);
    atomicAdd(o + 1, alpha * xv.y);
    atomicAdd(o + 2, alpha * xv.z);
    atomicAdd(o + 3, alpha * xv.w);
}

// ---------- BN + ReLU (with GAT bias) ----------
__global__ void bn_relu(const float* __restrict__ inb, const float* __restrict__ b,
                        const float* __restrict__ g, const float* __restrict__ be,
                        const float* __restrict__ mu, const float* __restrict__ var,
                        float* __restrict__ outb, int ctot, int n_items) {
    const int t = blockIdx.x * blockDim.x + threadIdx.x;
    if (t >= n_items) return;
    const int c = t % ctot;
    float v = inb[t] + b[c];
    v = (v - mu[c]) * rsqrtf(var[c] + BN_EPS) * g[c] + be[c];
    outb[t] = fmaxf(v, 0.f);
}

// ---------- classifier tail: logits = hc @ Wc2 + bc2; softmax ----------
__global__ void head_softmax(const float* __restrict__ hc, const float* __restrict__ Wc2,
                             const float* __restrict__ bc2, float* __restrict__ P) {
    const int n = blockIdx.x * blockDim.x + threadIdx.x;
    if (n >= NN) return;
    float lg[10];
    #pragma unroll
    for (int j = 0; j < 10; ++j) lg[j] = bc2[j];
    const float* hrow = hc + (size_t)n * 64;
    for (int c = 0; c < 64; ++c) {
        const float x = hrow[c];
        #pragma unroll
        for (int j = 0; j < 10; ++j) lg[j] += x * Wc2[c * 10 + j];
    }
    float mx = lg[0];
    #pragma unroll
    for (int j = 1; j < 10; ++j) mx = fmaxf(mx, lg[j]);
    float sum = 0.f;
    #pragma unroll
    for (int j = 0; j < 10; ++j) { lg[j] = __expf(lg[j] - mx); sum += lg[j]; }
    const float inv = 1.f / sum;
    #pragma unroll
    for (int j = 0; j < 10; ++j) P[(size_t)n * 10 + j] = lg[j] * inv;
}

// ---------- host-side layer driver ----------
static void gat_layer(const float* hin, int K, const float* W,
                      const float* asrc, const float* adst, const float* b,
                      const float* g, const float* be, const float* mu, const float* var,
                      int heads, const int* ei,
                      float* xp, float* es, float* ed, float* mbuf, float* sbuf,
                      float* wbuf, float* outb, float* hout, hipStream_t stream) {
    const int ctot = heads * 128;
    dim3 ggrid((NN + 127) / 128, ctot / 64);
    wmma_gemm<<<ggrid, 256, 0, stream>>>(hin, W, xp, NN, K, ctot, nullptr, 0);

    {
        int thr = NN * heads * 32;
        attn_scores<<<(thr + 255) / 256, 256, 0, stream>>>(xp, asrc, adst, es, ed, heads, NN * heads);
    }
    fill_f32<<<(NN * heads + 255) / 256, 256, 0, stream>>>(mbuf, -INFINITY, NN * heads);
    fill_f32<<<(NN * heads + 255) / 256, 256, 0, stream>>>(sbuf, 0.f, NN * heads);
    fill_f32<<<(NN * ctot + 255) / 256, 256, 0, stream>>>(outb, 0.f, NN * ctot);

    edge_max<<<(EP * heads + 255) / 256, 256, 0, stream>>>(ei, es, ed, mbuf, heads);
    edge_w<<<(EP * heads + 255) / 256, 256, 0, stream>>>(ei, es, ed, mbuf, sbuf, wbuf, heads);
    {
        int thr = EP * (ctot / 4);
        edge_agg<<<(thr + 255) / 256, 256, 0, stream>>>(ei, xp, wbuf, sbuf, outb, heads, ctot);
    }
    bn_relu<<<(NN * ctot + 255) / 256, 256, 0, stream>>>(outb, b, g, be, mu, var, hout, ctot, NN * ctot);
}

extern "C" void kernel_launch(void* const* d_in, const int* in_sizes, int n_in,
                              void* d_out, int out_size, void* d_ws, size_t ws_size,
                              hipStream_t stream) {
    const float* x  = (const float*)d_in[0];
    const int*   ei = (const int*)d_in[1];
    const float *W1 = (const float*)d_in[2],  *as1 = (const float*)d_in[3],  *ad1 = (const float*)d_in[4];
    const float *b1 = (const float*)d_in[5],  *g1 = (const float*)d_in[6],   *be1 = (const float*)d_in[7];
    const float *m1 = (const float*)d_in[8],  *v1 = (const float*)d_in[9];
    const float *W2 = (const float*)d_in[10], *as2 = (const float*)d_in[11], *ad2 = (const float*)d_in[12];
    const float *b2 = (const float*)d_in[13], *g2 = (const float*)d_in[14],  *be2 = (const float*)d_in[15];
    const float *m2 = (const float*)d_in[16], *v2 = (const float*)d_in[17];
    const float *W3 = (const float*)d_in[18], *as3 = (const float*)d_in[19], *ad3 = (const float*)d_in[20];
    const float *b3 = (const float*)d_in[21], *g3 = (const float*)d_in[22],  *be3 = (const float*)d_in[23];
    const float *m3 = (const float*)d_in[24], *v3 = (const float*)d_in[25];
    const float *Wc1 = (const float*)d_in[26], *bc1 = (const float*)d_in[27];
    const float *Wc2 = (const float*)d_in[28], *bc2 = (const float*)d_in[29];
    float* out = (float*)d_out;

    // workspace layout (floats)
    float* ws   = (float*)d_ws;
    float* hbuf = ws;                              // N*512
    float* xp   = hbuf + (size_t)NN * 512;         // N*512
    float* outb = xp   + (size_t)NN * 512;         // N*512
    float* es   = outb + (size_t)NN * 512;         // N*4
    float* ed   = es   + (size_t)NN * 4;           // N*4
    float* mbuf = ed   + (size_t)NN * 4;           // N*4
    float* sbuf = mbuf + (size_t)NN * 4;           // N*4
    float* wbuf = sbuf + (size_t)NN * 4;           // EP*4
    float* hc   = wbuf + (size_t)EP * 4;           // N*64

    // GAT layer 1: 256 -> 4x128 ; layer 2: 512 -> 4x128 ; layer 3: 512 -> 1x128
    gat_layer(x,    256, W1, as1, ad1, b1, g1, be1, m1, v1, 4, ei, xp, es, ed, mbuf, sbuf, wbuf, outb, hbuf, stream);
    gat_layer(hbuf, 512, W2, as2, ad2, b2, g2, be2, m2, v2, 4, ei, xp, es, ed, mbuf, sbuf, wbuf, outb, hbuf, stream);
    gat_layer(hbuf, 512, W3, as3, ad3, b3, g3, be3, m3, v3, 1, ei, xp, es, ed, mbuf, sbuf, wbuf, outb, out, stream);

    // classifier: relu(E_GNN @ Wc1 + bc1) -> hc ; logits/softmax -> P
    {
        dim3 ggrid((NN + 127) / 128, 1);
        wmma_gemm<<<ggrid, 256, 0, stream>>>(out, Wc1, hc, NN, 128, 64, bc1, 1);
    }
    head_softmax<<<(NN + 255) / 256, 256, 0, stream>>>(hc, Wc2, bc2, out + (size_t)NN * 128);
}
